// GaussianRenderer_37726992728447
// MI455X (gfx1250) — matrix operand order repaired and checked by reading, not compiled, not run on hardware
//
#include <hip/hip_runtime.h>

typedef __attribute__((ext_vector_type(16))) _Float16 v16h;
typedef __attribute__((ext_vector_type(8)))  float    v8f;

#define HH 256
#define WW 256
#define NG 1024
#define PLANE (HH*WW)
#define LOG2E 1.4426950408889634f

__global__ __launch_bounds__(256)
void gsplat_wmma_kernel(const float* __restrict__ means,
                        const float* __restrict__ colors,
                        const float* __restrict__ opacity,
                        const float* __restrict__ scales,
                        const float* __restrict__ Km,
                        const float* __restrict__ W2C,
                        float* __restrict__ out)
{
    // Per-gaussian record: {u, v, op_masked, log2e/(2*sigma^2)} -> one b128 broadcast load
    __shared__ float4 s_g[NG];                       // 16 KB
    __shared__ __align__(32) _Float16 s_B[16 * NG];  // 32 KB column-major B: [channel][gaussian]
    __shared__ float s_S[8][16][8];                  // 4 KB, per-wave result tile cols 0..4

    const int tid = threadIdx.x;

    // ---------- Phase 1: project gaussians, build B matrix in LDS ----------
    for (int g = tid; g < NG; g += 256) {
        float x  = means[g*3+0];
        float y  = means[g*3+1];
        float zw = means[g*3+2];
        float pc0 = W2C[0]*x + W2C[1]*y + W2C[2]*zw  + W2C[3];
        float pc1 = W2C[4]*x + W2C[5]*y + W2C[6]*zw  + W2C[7];
        float pc2 = W2C[8]*x + W2C[9]*y + W2C[10]*zw + W2C[11];
        float zc = pc2;
        float p0 = Km[0]*pc0 + Km[1]*pc1 + Km[2]*pc2;
        float p1 = Km[3]*pc0 + Km[4]*pc1 + Km[5]*pc2;
        float u = p0 / zc;
        float v = p1 / zc;
        bool valid = (zc > 1e-4f) && (u >= -(float)WW) && (u <= 2.0f*(float)WW)
                                  && (v >= -(float)HH) && (v <= 2.0f*(float)HH);
        float op = valid ? opacity[g] : 0.0f;
        if (!valid) { u = 0.0f; v = 0.0f; }
        float sig  = fmaxf(scales[g] * 256.0f, 1.0f);
        float inv2 = (0.5f * LOG2E) / (sig * sig);   // exp(-0.5 d2/s^2) == exp2(-d2*inv2)
        s_g[g] = make_float4(u, v, op, inv2);
        s_B[0*NG + g] = (_Float16)colors[g*3+0];
        s_B[1*NG + g] = (_Float16)colors[g*3+1];
        s_B[2*NG + g] = (_Float16)colors[g*3+2];
        s_B[3*NG + g] = (_Float16)(valid ? zc : 0.0f);
        s_B[4*NG + g] = (_Float16)1.0f;
        #pragma unroll
        for (int n = 5; n < 16; ++n) s_B[n*NG + g] = (_Float16)0.0f;
    }
    __syncthreads();

    // ---------- Phase 2: per-wave 16-pixel tile, WMMA K-loop ----------
    const int lane = tid & 31;
    const int wave = tid >> 5;
    const int py   = (int)blockIdx.x >> 1;
    const int px0  = ((int)blockIdx.x & 1) * 128 + wave * 16;

    const int m   = lane & 15;      // A-matrix row M == pixel within tile, also B column N
    const int hi  = lane >> 4;      // lane half selects K sub-range
    const int kA  = hi * 8;
    const int kB  = hi * 16;
    const float fx = (float)(px0 + m);
    const float fy = (float)py;

    v8f acc = {};
    for (int base = 0; base < NG; base += 32) {
        v16h a;
        #pragma unroll
        for (int j = 0; j < 8; ++j) {        // K = base + kB + j   -> A VGPRs 0..3
            float4 gd = s_g[base + kB + j];
            float dx = fx - gd.x, dy = fy - gd.y;
            float al = gd.z * __builtin_amdgcn_exp2f(-(dx*dx + dy*dy) * gd.w);
            a[j] = (_Float16)fminf(al, 0.95f);
        }
        #pragma unroll
        for (int j = 0; j < 8; ++j) {        // K = base + kB + 8 + j -> A VGPRs 4..7
            float4 gd = s_g[base + kB + 8 + j];
            float dx = fx - gd.x, dy = fy - gd.y;
            float al = gd.z * __builtin_amdgcn_exp2f(-(dx*dx + dy*dy) * gd.w);
            a[8 + j] = (_Float16)fminf(al, 0.95f);
        }
        // B: K = base+kB .. base+kB+15 are contiguous halves in s_B (2x ds_load_b128)
        v16h b = *(const v16h*)&s_B[m*NG + base + kB];

        acc = __builtin_amdgcn_wmma_f32_16x16x32_f16(
                  false, a, false, b, (short)0, acc, false, false);
    }

    // stash columns 0..4 of this wave's 16x16 f32 tile (rows: lanes0-15 -> 0..7, lanes16-31 -> 8..15)
    if (m < 5) {
        #pragma unroll
        for (int vv = 0; vv < 8; ++vv)
            s_S[wave][hi*8 + vv][m] = acc[vv];
    }
    __syncthreads();

    // ---------- Phase 3: finalize 128 pixels ----------
    if (tid < 128) {
        int wsrc = tid >> 4, mm = tid & 15;
        float S0 = s_S[wsrc][mm][0];
        float S1 = s_S[wsrc][mm][1];
        float S2 = s_S[wsrc][mm][2];
        float S3 = s_S[wsrc][mm][3];
        float S4 = s_S[wsrc][mm][4];
        float invden = __builtin_amdgcn_rcpf(S4 + 1e-6f);
        float accum  = fminf(fmaxf(S4, 0.0f), 1.0f);
        float bgm    = 1.0f - accum;
        int x   = ((int)blockIdx.x & 1) * 128 + tid;
        int idx = py * WW + x;
        out[0*PLANE + idx] = fminf(fmaxf(S0*invden*accum + bgm, 0.0f), 1.0f);
        out[1*PLANE + idx] = fminf(fmaxf(S1*invden*accum + bgm, 0.0f), 1.0f);
        out[2*PLANE + idx] = fminf(fmaxf(S2*invden*accum + bgm, 0.0f), 1.0f);
        out[3*PLANE + idx] = accum;
        out[4*PLANE + idx] = S3 * invden;
    }
}

extern "C" void kernel_launch(void* const* d_in, const int* in_sizes, int n_in,
                              void* d_out, int out_size, void* d_ws, size_t ws_size,
                              hipStream_t stream) {
    (void)in_sizes; (void)n_in; (void)out_size; (void)d_ws; (void)ws_size;
    const float* means   = (const float*)d_in[0];
    const float* colors  = (const float*)d_in[1];
    const float* opacity = (const float*)d_in[2];
    const float* scales  = (const float*)d_in[3];
    const float* Km      = (const float*)d_in[4];
    const float* W2C     = (const float*)d_in[5];
    float* out = (float*)d_out;

    dim3 grid(512), block(256);
    hipLaunchKernelGGL(gsplat_wmma_kernel, grid, block, 0, stream,
                       means, colors, opacity, scales, Km, W2C, out);
}
